// SDPA_17763984736846
// MI455X (gfx1250) — matrix operand order repaired and checked by reading, not compiled, not run on hardware
//
#include <hip/hip_runtime.h>
#include <hip/hip_bf16.h>

// MI455X (gfx1250) sliding-window attention with f16 WMMA (f32 accum) and
// CDNA5 async global->LDS staging (GLOBAL_LOAD_ASYNC_TO_LDS_B128 / ASYNCcnt).
// Workspace layout (f16): xh[4096*1024] | wh[3072*1024] | Q|K|V [2*16*2048*64] each
// Total ~39.8 MB.

typedef __attribute__((ext_vector_type(16))) _Float16 v16h;
typedef __attribute__((ext_vector_type(8)))  _Float16 v8h;
typedef __attribute__((ext_vector_type(2)))  _Float16 v2h;
typedef __attribute__((ext_vector_type(8)))  float    v8f;

#define S_LEN  2048
#define DMODEL 1024
#define NHEAD  16
#define HDIM   64
#define BATCH  2
#define M_TOT  (BATCH * S_LEN)   // 4096
#define N_TOT  (3 * DMODEL)      // 3072
#define K_TOT  DMODEL            // 1024

// ---- WMMA f16 fragment index helpers (per CDNA5 ISA 7.12.2) ----
// A 16x32: lane<16 -> K {0..7, 16..23}; lane>=16 -> K {8..15, 24..31}; M = lane%16
__device__ __forceinline__ int a_k(int lane, int v) {
  return ((v & 4) << 2) + ((lane & 16) >> 1) + ((v & 3) << 1);
}
// B 32x16: lane<16 -> K {0..7, 16..23}; lane>=16 -> K {8..15, 24..31}; N = lane%16
__device__ __forceinline__ int b_k(int lane, int v) {
  return ((v & 4) << 2) + ((lane & 16) >> 1) + ((v & 3) << 1);
}

__device__ __forceinline__ unsigned lds_off(const void* p) {
  // Generic shared pointer: high dword = shared aperture, low dword = LDS offset.
  return (unsigned)(unsigned long long)(uintptr_t)p;
}

__global__ void cvt_f32_f16(const float* __restrict__ in, _Float16* __restrict__ out, int n) {
  int i = blockIdx.x * blockDim.x + threadIdx.x;
  if (i < n) out[i] = (_Float16)in[i];
}

// ---------------- QKV projection GEMM: [4096,1024] x [1024,3072] ----------------
__global__ __launch_bounds__(256) void qkv_gemm_wmma(
    const _Float16* __restrict__ xh, const _Float16* __restrict__ wh,
    const float* __restrict__ bias,
    _Float16* __restrict__ Qb, _Float16* __restrict__ Kb, _Float16* __restrict__ Vb)
{
  __shared__ __align__(16) _Float16 As[2][128][40];  // double-buffered 128x32 tiles
  __shared__ __align__(16) _Float16 Bs[2][64][40];

  const int tid  = threadIdx.x;
  const int lane = tid & 31;
  const int wave = tid >> 5;
  const int m0 = blockIdx.y * 128;
  const int n0 = blockIdx.x * 64;
  const int wm = (wave & 3) * 32;   // wave tile 32x32
  const int wn = (wave >> 2) * 32;

  v8f acc00 = {}, acc01 = {}, acc10 = {}, acc11 = {};

  const int arow = tid >> 1, apart = (tid & 1) * 16;  // 128 rows x 2 chunks of 16 halfs
  const int brow = tid >> 2, bpart = (tid & 3) * 8;   // 64 rows  x 4 chunks of 8 halfs

  // Async-stage one K-slice (3 b128 copies per thread) into buffer `buf`.
  auto stage = [&](int buf, int kk) {
    const _Float16* ga = xh + (size_t)(m0 + arow) * K_TOT + kk + apart;
    const _Float16* gb = wh + (size_t)(n0 + brow) * K_TOT + kk + bpart;
    unsigned la = lds_off(&As[buf][arow][apart]);
    unsigned lb = lds_off(&Bs[buf][brow][bpart]);
    asm volatile(
        "global_load_async_to_lds_b128 %0, %2, off\n\t"
        "global_load_async_to_lds_b128 %0, %2, off offset:16\n\t"
        "global_load_async_to_lds_b128 %1, %3, off"
        :: "v"(la), "v"(lb),
           "v"((unsigned long long)(uintptr_t)ga),
           "v"((unsigned long long)(uintptr_t)gb)
        : "memory");
  };

  stage(0, 0);
  for (int kk = 0; kk < K_TOT; kk += 32) {
    const int buf = (kk >> 5) & 1;
    if (kk + 32 < K_TOT) {
      stage(buf ^ 1, kk + 32);
      // <=3 outstanding => current buffer's 3 copies have completed
      asm volatile("s_wait_asynccnt 0x3" ::: "memory");
    } else {
      asm volatile("s_wait_asynccnt 0x0" ::: "memory");
    }
    __syncthreads();   // all waves' current-buffer copies complete

    v16h a0, a1, b0, b1;
#pragma unroll
    for (int v = 0; v < 8; ++v) {
      const int ka = a_k(lane, v);
      v2h p0 = *reinterpret_cast<const v2h*>(&As[buf][wm + (lane & 15)][ka]);
      v2h p1 = *reinterpret_cast<const v2h*>(&As[buf][wm + 16 + (lane & 15)][ka]);
      a0[2*v] = p0[0]; a0[2*v+1] = p0[1];
      a1[2*v] = p1[0]; a1[2*v+1] = p1[1];
      const int kb = b_k(lane, v);
      v2h q0 = *reinterpret_cast<const v2h*>(&Bs[buf][wn + (lane & 15)][kb]);
      v2h q1 = *reinterpret_cast<const v2h*>(&Bs[buf][wn + 16 + (lane & 15)][kb]);
      b0[2*v] = q0[0]; b0[2*v+1] = q0[1];
      b1[2*v] = q1[0]; b1[2*v+1] = q1[1];
    }
    acc00 = __builtin_amdgcn_wmma_f32_16x16x32_f16(false, a0, false, b0, (short)0, acc00, false, false);
    acc01 = __builtin_amdgcn_wmma_f32_16x16x32_f16(false, a0, false, b1, (short)0, acc01, false, false);
    acc10 = __builtin_amdgcn_wmma_f32_16x16x32_f16(false, a1, false, b0, (short)0, acc10, false, false);
    acc11 = __builtin_amdgcn_wmma_f32_16x16x32_f16(false, a1, false, b1, (short)0, acc11, false, false);

    __syncthreads();   // keep next iteration's async writes off the live buffer
  }

  // Epilogue: add bias, route to Q/K/V head-major f16 buffers [B,H,S,64]
#pragma unroll
  for (int i = 0; i < 2; ++i) {
#pragma unroll
    for (int j = 0; j < 2; ++j) {
      v8f acc = (i == 0) ? (j == 0 ? acc00 : acc01) : (j == 0 ? acc10 : acc11);
#pragma unroll
      for (int r = 0; r < 8; ++r) {
        const int row = m0 + wm + i * 16 + r + ((lane & 16) ? 8 : 0);
        const int col = n0 + wn + j * 16 + (lane & 15);
        const float val = acc[r] + bias[col];
        const int bb = row >> 11;       // / 2048
        const int s  = row & 2047;
        const int which = col >> 10;    // 0=Q, 1=K, 2=V
        const int f = col & 1023;
        const int hh = f >> 6, d = f & 63;
        const size_t idx = (((size_t)(bb * NHEAD + hh) * S_LEN) + s) * HDIM + d;
        const _Float16 hv = (_Float16)val;
        if (which == 0) Qb[idx] = hv;
        else if (which == 1) Kb[idx] = hv;
        else Vb[idx] = hv;
      }
    }
  }
}

// ---------------- Sliding-window attention, one wave per (b, h, 16-query tile) ----------------
__global__ __launch_bounds__(128) void swa_attn_wmma(
    const _Float16* __restrict__ Qb, const _Float16* __restrict__ Kb,
    const _Float16* __restrict__ Vb, float* __restrict__ out)
{
  __shared__ __align__(16) _Float16 Ks[4][32][72];    // per-wave 32x64 K tile (144B rows)
  __shared__ __align__(16) _Float16 Vs[4][32][72];    // per-wave 32x64 V tile
  __shared__ __align__(16) _Float16 Pbuf[4][16][34];  // per-wave P transpose buffer

  const int lane = threadIdx.x & 31;
  const int wave = threadIdx.x >> 5;
  const int job  = blockIdx.x * 4 + wave;   // 4096 jobs = 2*16*128
  const int qb = job & 127;
  const int h  = (job >> 7) & (NHEAD - 1);
  const int b  = job >> 11;
  const int q0 = qb * 16;

  const size_t head_off = (size_t)(b * NHEAD + h) * S_LEN * HDIM;
  const _Float16* Qh = Qb + head_off;
  const _Float16* Kh = Kb + head_off;
  const _Float16* Vh = Vb + head_off;

  // Q fragments (A layout), head dim 64 -> two K=32 fragments
  v16h qf0, qf1;
  {
    const _Float16* qrow = Qh + (size_t)(q0 + (lane & 15)) * HDIM;
#pragma unroll
    for (int v = 0; v < 8; ++v) {
      const int k = a_k(lane, v);
      v2h p0 = *reinterpret_cast<const v2h*>(qrow + k);
      v2h p1 = *reinterpret_cast<const v2h*>(qrow + 32 + k);
      qf0[2*v] = p0[0]; qf0[2*v+1] = p0[1];
      qf1[2*v] = p1[0]; qf1[2*v+1] = p1[1];
    }
  }

  v8f o0 = {}, o1 = {}, o2 = {}, o3 = {};
  float mrow[8], lrow[8];
#pragma unroll
  for (int r = 0; r < 8; ++r) { mrow[r] = -1e30f; lrow[r] = 0.0f; }

  // Window [-127,+128] for 16 queries -> 18 key tiles of 16 = 9 pairs of 32 keys
  for (int p = 0; p < 9; ++p) {
    const int kb0 = q0 - 128 + p * 32;
    if (kb0 + 31 < 0 || kb0 >= S_LEN) continue;   // wave-uniform skip

    // ---- async-stage K and V tiles: one 128B row per lane, 8 b128 copies each ----
    {
      const int kc = min(max(kb0 + lane, 0), S_LEN - 1);
      unsigned lk = lds_off(&Ks[wave][lane][0]);
      unsigned lv = lds_off(&Vs[wave][lane][0]);
      unsigned long long ak = (unsigned long long)(uintptr_t)(Kh + (size_t)kc * HDIM);
      unsigned long long av = (unsigned long long)(uintptr_t)(Vh + (size_t)kc * HDIM);
      asm volatile(
          "global_load_async_to_lds_b128 %0, %2, off\n\t"
          "global_load_async_to_lds_b128 %0, %2, off offset:16\n\t"
          "global_load_async_to_lds_b128 %0, %2, off offset:32\n\t"
          "global_load_async_to_lds_b128 %0, %2, off offset:48\n\t"
          "global_load_async_to_lds_b128 %0, %2, off offset:64\n\t"
          "global_load_async_to_lds_b128 %0, %2, off offset:80\n\t"
          "global_load_async_to_lds_b128 %0, %2, off offset:96\n\t"
          "global_load_async_to_lds_b128 %0, %2, off offset:112\n\t"
          "global_load_async_to_lds_b128 %1, %3, off\n\t"
          "global_load_async_to_lds_b128 %1, %3, off offset:16\n\t"
          "global_load_async_to_lds_b128 %1, %3, off offset:32\n\t"
          "global_load_async_to_lds_b128 %1, %3, off offset:48\n\t"
          "global_load_async_to_lds_b128 %1, %3, off offset:64\n\t"
          "global_load_async_to_lds_b128 %1, %3, off offset:80\n\t"
          "global_load_async_to_lds_b128 %1, %3, off offset:96\n\t"
          "global_load_async_to_lds_b128 %1, %3, off offset:112"
          :: "v"(lk), "v"(lv), "v"(ak), "v"(av)
          : "memory");
      asm volatile("s_wait_asynccnt 0x0" ::: "memory");  // per-wave: all 16 copies done
      __builtin_amdgcn_wave_barrier();
    }

    // ---- S[16x32] = Q(16x64) @ K^T(64x32): 2 key tiles x 2 K-halves ----
    v8f sc0 = {}, sc1 = {};
#pragma unroll
    for (int j = 0; j < 2; ++j) {
      const _Float16* krow = &Ks[wave][j * 16 + (lane & 15)][0];
      v16h bk0, bk1;
#pragma unroll
      for (int v = 0; v < 8; ++v) {
        const int k = b_k(lane, v);
        v2h p0 = *reinterpret_cast<const v2h*>(krow + k);
        v2h p1 = *reinterpret_cast<const v2h*>(krow + 32 + k);
        bk0[2*v] = p0[0]; bk0[2*v+1] = p0[1];
        bk1[2*v] = p1[0]; bk1[2*v+1] = p1[1];
      }
      v8f s = {};
      s = __builtin_amdgcn_wmma_f32_16x16x32_f16(false, qf0, false, bk0, (short)0, s, false, false);
      s = __builtin_amdgcn_wmma_f32_16x16x32_f16(false, qf1, false, bk1, (short)0, s, false, false);
      if (j == 0) sc0 = s; else sc1 = s;
    }

    // ---- mask + online softmax (row reductions across 16-lane halves) ----
    const float scale = 0.125f;   // 1/sqrt(64)
    float alpha[8];
#pragma unroll
    for (int r = 0; r < 8; ++r) {
      const int row  = q0 + r + ((lane & 16) ? 8 : 0);
      const int key0 = kb0 + (lane & 15);
      const int key1 = key0 + 16;
      const bool ok0 = (key0 >= 0) && (key0 < S_LEN) && (key0 - row <= 128) && (row - key0 <= 127);
      const bool ok1 = (key1 >= 0) && (key1 < S_LEN) && (key1 - row <= 128) && (row - key1 <= 127);
      float s0 = ok0 ? sc0[r] * scale : -1e30f;
      float s1 = ok1 ? sc1[r] * scale : -1e30f;
      float best = fmaxf(s0, s1);
#pragma unroll
      for (int m = 8; m >= 1; m >>= 1) best = fmaxf(best, __shfl_xor(best, m, 32));
      const float mnew = fmaxf(mrow[r], best);
      const float a  = __expf(mrow[r] - mnew);
      const float p0 = __expf(s0 - mnew);
      const float p1 = __expf(s1 - mnew);
      float ps = p0 + p1;
#pragma unroll
      for (int m = 8; m >= 1; m >>= 1) ps += __shfl_xor(ps, m, 32);
      lrow[r] = lrow[r] * a + ps;
      mrow[r] = mnew;
      alpha[r] = a;
      sc0[r] = p0; sc1[r] = p1;
    }
#pragma unroll
    for (int r = 0; r < 8; ++r) {
      o0[r] *= alpha[r]; o1[r] *= alpha[r]; o2[r] *= alpha[r]; o3[r] *= alpha[r];
    }

    // ---- transpose P (C layout -> A layout) through per-wave LDS ----
    __builtin_amdgcn_wave_barrier();
#pragma unroll
    for (int r = 0; r < 8; ++r) {
      const int prow = r + ((lane & 16) ? 8 : 0);
      Pbuf[wave][prow][lane & 15]        = (_Float16)sc0[r];
      Pbuf[wave][prow][16 + (lane & 15)] = (_Float16)sc1[r];
    }
    __builtin_amdgcn_wave_barrier();
    v16h pf;
#pragma unroll
    for (int v = 0; v < 8; ++v) {
      const int k = a_k(lane, v);
      v2h pp = *reinterpret_cast<const v2h*>(&Pbuf[wave][lane & 15][k]);
      pf[2*v] = pp[0]; pf[2*v+1] = pp[1];
    }

    // ---- O(16x64) += P(16x32) @ V(32x64) : 4 hd tiles, V from LDS ----
#pragma unroll
    for (int t = 0; t < 4; ++t) {
      v16h vf;
#pragma unroll
      for (int v = 0; v < 8; ++v) {
        const int k = b_k(lane, v);   // key offset within pair
        vf[2*v]   = Vs[wave][k][t * 16 + (lane & 15)];
        vf[2*v+1] = Vs[wave][k + 1][t * 16 + (lane & 15)];
      }
      v8f& o = (t == 0) ? o0 : (t == 1) ? o1 : (t == 2) ? o2 : o3;
      o = __builtin_amdgcn_wmma_f32_16x16x32_f16(false, pf, false, vf, (short)0, o, false, false);
    }
    __builtin_amdgcn_wave_barrier();   // done reading Ks/Vs before next pair's asyncs
  }

  // ---- normalize and store fp32 output [B, S, H*hd] ----
#pragma unroll
  for (int r = 0; r < 8; ++r) {
    const int row = q0 + r + ((lane & 16) ? 8 : 0);
    const float inv = 1.0f / lrow[r];
    float* orow = out + ((size_t)b * S_LEN + row) * DMODEL + h * HDIM + (lane & 15);
    orow[0]  = o0[r] * inv;
    orow[16] = o1[r] * inv;
    orow[32] = o2[r] * inv;
    orow[48] = o3[r] * inv;
  }
}

extern "C" void kernel_launch(void* const* d_in, const int* in_sizes, int n_in,
                              void* d_out, int out_size, void* d_ws, size_t ws_size,
                              hipStream_t stream) {
  const float* x    = (const float*)d_in[0];
  const float* Wq   = (const float*)d_in[1];
  const float* bias = (const float*)d_in[2];
  float* out = (float*)d_out;

  _Float16* xh = (_Float16*)d_ws;
  _Float16* wh = xh + (size_t)M_TOT * K_TOT;                    // +4M halfs
  _Float16* Qb = wh + (size_t)N_TOT * K_TOT;                    // +3M halfs
  _Float16* Kb = Qb + (size_t)BATCH * NHEAD * S_LEN * HDIM;     // +4M halfs
  _Float16* Vb = Kb + (size_t)BATCH * NHEAD * S_LEN * HDIM;     // +4M halfs

  const int nx = M_TOT * K_TOT;   // 4,194,304
  const int nw = N_TOT * K_TOT;   // 3,145,728
  cvt_f32_f16<<<nx / 256, 256, 0, stream>>>(x, xh, nx);
  cvt_f32_f16<<<nw / 256, 256, 0, stream>>>(Wq, wh, nw);
  qkv_gemm_wmma<<<dim3(N_TOT / 64, M_TOT / 128), 256, 0, stream>>>(xh, wh, bias, Qb, Kb, Vb);
  swa_attn_wmma<<<(BATCH * NHEAD * (S_LEN / 16)) / 4, 128, 0, stream>>>(Qb, Kb, Vb, out);
}
